// OptimusPrime_747324310349
// MI455X (gfx1250) — matrix operand order repaired and checked by reading, not compiled, not run on hardware
//
#include <hip/hip_runtime.h>
#include <math.h>

typedef __attribute__((ext_vector_type(16))) _Float16 v16h;
typedef __attribute__((ext_vector_type(8)))  _Float16 v8h;
typedef __attribute__((ext_vector_type(8)))  float    v8f;
typedef _Float16 h16;

#define NW        8              // waves per block (256 threads, wave32)
#define NPAIR     4              // wave-pairs (groups) per block
#define NB        16             // batch elements per group
#define SEQ       11
#define TOK       (NB * SEQ)     // 176 token rows = 11 exact 16-row M-tiles
#define DMODEL    32
#define FFDIM     128
#define MTILES    11

// ---------------- LDS layout (bytes) ----------------
#define OFF_QKVW  0                               // 2 x [96][32] f16
#define OFF_OUTW  (OFF_QKVW + 2*96*32*2)          // 2 x [32][32] f16
#define OFF_W1    (OFF_OUTW + 2*32*32*2)          // 2 x [128][32] f16
#define OFF_W2    (OFF_W1   + 2*128*32*2)         // 2 x [32][128] f16
#define OFF_QKVB  (OFF_W2   + 2*32*128*2)         // 2 x [96] f32
#define OFF_OUTB  (OFF_QKVB + 2*96*4)
#define OFF_B1    (OFF_OUTB + 2*32*4)
#define OFF_B2    (OFF_B1   + 2*128*4)
#define OFF_LNAG  (OFF_B2   + 2*32*4)
#define OFF_LNAB  (OFF_LNAG + 2*32*4)
#define OFF_LNBG  (OFF_LNAB + 2*32*4)
#define OFF_LNBB  (OFF_LNBG + 2*32*4)
#define OFF_WF    (OFF_LNBB + 2*32*4)             // [10][32] f32
#define OFF_BF    (OFF_WF   + 320*4)
#define OFF_CLS   (OFF_BF   + 320*4)
#define OFF_CLFW  (OFF_CLS  + 32*4)
#define OFF_CLFB  (OFF_CLFW + 32*4)
#define OFF_GRP   55552                           // 256B-aligned start of group areas
#define GOFF_ACT  0                               // [176][32] f32 residual stream
#define GOFF_QKV  (TOK*DMODEL*4)                  // [176][96] f16 (FFN scratch reuse)
#define GOFF_CTX  (GOFF_QKV + TOK*96*2)           // [176][32] f16 attention output
#define GSTRIDE   (GOFF_CTX + TOK*DMODEL*2)       // 67584 per group
#define SMEM_BYTES (OFF_GRP + NPAIR*GSTRIDE)      // 325888 <= 320KB

#define WMMA(a,b,c) __builtin_amdgcn_wmma_f32_16x16x32_f16(false,(a),false,(b),(short)0,(c),false,false)
#define LDSFENCE() asm volatile("s_wait_dscnt 0x0" ::: "memory")

struct BlkW {
    const float *qkvw, *qkvb, *outw, *outb, *lnag, *lnab;
    const float *w1, *b1, *w2, *b2, *lnbg, *lnbb;
};
struct TParams {
    const float *x, *wf, *bf, *cls;
    BlkW blk[2];
    const float *clfw, *clfb;
    float* out;
    int Btot;
    int ngroups;     // ceil(Btot/NB)
    int ngblocks;    // ceil(ngroups/NPAIR)
};
struct WP { // resolved LDS weight pointers for one transformer block
    const h16 *qkvw, *outw, *w1, *w2;
    const float *qkvb, *outb, *b1, *b2, *lnag, *lnab, *lnbg, *lnbb;
};

// ---- fragment loaders (layouts per CDNA5 ISA 7.12.2, wave32) ----
// B 32x16 f16 from row-major [N][K]: lane<16 -> N=lane, K 0..7,16..23 ; lane>=16 -> N=lane-16, K 8..15,24..31
static __device__ __forceinline__ v16h frag_b(const h16* w, int ldk, int n, int kb) {
    const v8h* p = (const v8h*)(w + n * ldk + kb - ((kb & 16) >> 1));
    v8h lo = p[0], hi = p[2];
    v16h r;
#pragma unroll
    for (int i = 0; i < 8; ++i) { r[i] = lo[i]; r[i + 8] = hi[i]; }
    return r;
}
// A 16x32 f16 slice: elems 0..7 = K kb..kb+7, elems 8..15 = K kb+16..kb+23
static __device__ __forceinline__ v16h frag_a16(const h16* m, int ld, int row, int kb) {
    const h16* p = m + row * ld + kb;
    v8h lo = *(const v8h*)p;
    v8h hi = *(const v8h*)(p + 16);
    v16h r;
#pragma unroll
    for (int i = 0; i < 8; ++i) { r[i] = lo[i]; r[i + 8] = hi[i]; }
    return r;
}
// A 16x32 built from f32 LDS with on-the-fly f16 convert
static __device__ __forceinline__ v16h frag_a32(const float* m, int ld, int row, int kb) {
    const float* p = m + row * ld + kb;
    v16h r;
#pragma unroll
    for (int i = 0; i < 8; ++i) { r[i] = (_Float16)p[i]; r[i + 8] = (_Float16)p[i + 16]; }
    return r;
}

static __device__ __forceinline__ void lds_layernorm(float* ACT, const float* g, const float* b, int id) {
    for (int t = id; t < TOK; t += 64) {
        float* row = ACT + t * DMODEL;
        float mu = 0.f;
#pragma unroll
        for (int d = 0; d < DMODEL; ++d) mu += row[d];
        mu *= (1.0f / DMODEL);
        float var = 0.f;
#pragma unroll
        for (int d = 0; d < DMODEL; ++d) { float z = row[d] - mu; var += z * z; }
        var *= (1.0f / DMODEL);
        float inv = rsqrtf(var + 1e-5f);
#pragma unroll
        for (int d = 0; d < DMODEL; ++d) row[d] = (row[d] - mu) * inv * g[d] + b[d];
    }
}

// One transformer block for a 16-batch group, cooperatively run by a wave pair.
static __device__ void run_block(float* ACT, h16* QKV, h16* CTX, const WP& W,
                                 int lane, int pw, bool gvalid, int nvalid) {
    const int li   = lane & 15;
    const int rofs = (lane < 16) ? 0 : 8;   // A K-base low == D row offset
    const int kbB  = rofs * 2;              // B K-base (0 or 16)
    const int id   = pw * 32 + lane;        // 0..63 within the pair

    // ---------- QKV projection: [176x32] x [32x96] + bias -> QKV f16 ----------
    if (gvalid) {
#pragma unroll 1
        for (int mt = pw; mt < MTILES; mt += 2) {
            v16h a = frag_a32(ACT, DMODEL, mt * 16 + li, rofs);
#pragma unroll
            for (int nt = 0; nt < 6; ++nt) {
                v16h b = frag_b(W.qkvw, DMODEL, nt * 16 + li, kbB);
                float bias = W.qkvb[nt * 16 + li];
                v8f c;
#pragma unroll
                for (int i = 0; i < 8; ++i) c[i] = bias;
                c = WMMA(a, b, c);
                int n = nt * 16 + li;
#pragma unroll
                for (int r = 0; r < 8; ++r)
                    QKV[(mt * 16 + rofs + r) * 96 + n] = (_Float16)c[r];
            }
        }
    }
    __syncthreads();

    // ---------- attention (VALU): pair-lane = 16 batches x 4 heads ----------
    {
        int bl = id & 15, hh = id >> 4;
        if (gvalid && bl < nvalid) {
            int rb = bl * SEQ;
#pragma unroll 1
            for (int qs = 0; qs < SEQ; ++qs) {
                v8h qv = *(const v8h*)(QKV + (rb + qs) * 96 + hh * 8);
                float qf[8];
#pragma unroll
                for (int d = 0; d < 8; ++d) qf[d] = (float)qv[d] * 0.35355339059f; // 1/sqrt(8)
                float sc[SEQ];
                float mx = -3.0e38f;
#pragma unroll
                for (int ks = 0; ks < SEQ; ++ks) {
                    v8h kv = *(const v8h*)(QKV + (rb + ks) * 96 + 32 + hh * 8);
                    float s = 0.f;
#pragma unroll
                    for (int d = 0; d < 8; ++d) s += qf[d] * (float)kv[d];
                    sc[ks] = s;
                    mx = fmaxf(mx, s);
                }
                float sum = 0.f, acc[8] = {0, 0, 0, 0, 0, 0, 0, 0};
#pragma unroll
                for (int ks = 0; ks < SEQ; ++ks) {
                    float p = __expf(sc[ks] - mx);
                    sum += p;
                    v8h vv = *(const v8h*)(QKV + (rb + ks) * 96 + 64 + hh * 8);
#pragma unroll
                    for (int d = 0; d < 8; ++d) acc[d] += p * (float)vv[d];
                }
                float inv = 1.0f / sum;
#pragma unroll
                for (int d = 0; d < 8; ++d)
                    CTX[(rb + qs) * DMODEL + hh * 8 + d] = (_Float16)(acc[d] * inv);
            }
        }
    }
    __syncthreads();

    // ---------- out-proj: [176x32] x [32x32] + bias, residual into ACT ----------
    if (gvalid) {
#pragma unroll 1
        for (int mt = pw; mt < MTILES; mt += 2) {
            v16h a = frag_a16(CTX, DMODEL, mt * 16 + li, rofs);
#pragma unroll
            for (int nt = 0; nt < 2; ++nt) {
                v16h b = frag_b(W.outw, DMODEL, nt * 16 + li, kbB);
                float bias = W.outb[nt * 16 + li];
                v8f c;
#pragma unroll
                for (int i = 0; i < 8; ++i) c[i] = bias;
                c = WMMA(a, b, c);
                int n = nt * 16 + li;
#pragma unroll
                for (int r = 0; r < 8; ++r)
                    ACT[(mt * 16 + rofs + r) * DMODEL + n] += c[r];
            }
        }
    }
    __syncthreads();
    if (gvalid) lds_layernorm(ACT, W.lnag, W.lnab, id);
    __syncthreads();

    // ---------- FFN: x@w1^T -> GELU -> @w2^T, residual into ACT ----------
    h16* HS = QKV + pw * (16 * FFDIM);   // private per-wave [16][128] scratch in dead QKV region
    if (gvalid) {
#pragma unroll 1
        for (int mt = pw; mt < MTILES; mt += 2) {
            v16h a = frag_a32(ACT, DMODEL, mt * 16 + li, rofs);
#pragma unroll
            for (int nt = 0; nt < 8; ++nt) {
                v16h b = frag_b(W.w1, DMODEL, nt * 16 + li, kbB);
                float bias = W.b1[nt * 16 + li];
                v8f c;
#pragma unroll
                for (int i = 0; i < 8; ++i) c[i] = bias;
                c = WMMA(a, b, c);
                int n = nt * 16 + li;
#pragma unroll
                for (int r = 0; r < 8; ++r) {
                    float xg = c[r];
                    float gl = 0.5f * xg * (1.0f + erff(xg * 0.70710678f)); // exact GELU
                    HS[(rofs + r) * FFDIM + n] = (_Float16)gl;
                }
            }
            LDSFENCE();   // within-wave: HS stores -> HS fragment loads
#pragma unroll
            for (int nt = 0; nt < 2; ++nt) {
                float bias = W.b2[nt * 16 + li];
                v8f c;
#pragma unroll
                for (int i = 0; i < 8; ++i) c[i] = bias;
#pragma unroll
                for (int kc = 0; kc < 4; ++kc) { // K = 128 in 4 chained WMMAs
                    v16h a2 = frag_a16(HS, FFDIM, li, kc * 32 + rofs);
                    v16h b2 = frag_b(W.w2, FFDIM, nt * 16 + li, kc * 32 + kbB);
                    c = WMMA(a2, b2, c);
                }
                int n = nt * 16 + li;
#pragma unroll
                for (int r = 0; r < 8; ++r)
                    ACT[(mt * 16 + rofs + r) * DMODEL + n] += c[r];
            }
            LDSFENCE();
        }
    }
    __syncthreads();
    if (gvalid) lds_layernorm(ACT, W.lnbg, W.lnbb, id);
    __syncthreads();
}

static __device__ __forceinline__ void stage_cvt(const float* s, h16* d, int n, int tid) {
    for (int i = tid; i < n; i += 256) d[i] = (_Float16)s[i];
}
static __device__ __forceinline__ void stage_cpy(const float* s, float* d, int n, int tid) {
    for (int i = tid; i < n; i += 256) d[i] = s[i];
}

__global__ void __launch_bounds__(256)
fttrans_fused_kernel(TParams P) {
    extern __shared__ char sm[];
    const int tid = threadIdx.x;

    // ---- stage weights to LDS (f32 -> f16 for GEMM operands) ----
    h16*   qkvw_s = (h16*)(sm + OFF_QKVW);
    h16*   outw_s = (h16*)(sm + OFF_OUTW);
    h16*   w1_s   = (h16*)(sm + OFF_W1);
    h16*   w2_s   = (h16*)(sm + OFF_W2);
    float* qkvb_s = (float*)(sm + OFF_QKVB);
    float* outb_s = (float*)(sm + OFF_OUTB);
    float* b1_s   = (float*)(sm + OFF_B1);
    float* b2_s   = (float*)(sm + OFF_B2);
    float* lnag_s = (float*)(sm + OFF_LNAG);
    float* lnab_s = (float*)(sm + OFF_LNAB);
    float* lnbg_s = (float*)(sm + OFF_LNBG);
    float* lnbb_s = (float*)(sm + OFF_LNBB);
    float* wf_s   = (float*)(sm + OFF_WF);
    float* bf_s   = (float*)(sm + OFF_BF);
    float* cls_s  = (float*)(sm + OFF_CLS);
    float* clfw_s = (float*)(sm + OFF_CLFW);
    float* clfb_s = (float*)(sm + OFF_CLFB);

#pragma unroll
    for (int b = 0; b < 2; ++b) {
        stage_cvt(P.blk[b].qkvw, qkvw_s + b * 3072, 3072, tid);
        stage_cvt(P.blk[b].outw, outw_s + b * 1024, 1024, tid);
        stage_cvt(P.blk[b].w1,   w1_s   + b * 4096, 4096, tid);
        stage_cvt(P.blk[b].w2,   w2_s   + b * 4096, 4096, tid);
        stage_cpy(P.blk[b].qkvb, qkvb_s + b * 96,  96,  tid);
        stage_cpy(P.blk[b].outb, outb_s + b * 32,  32,  tid);
        stage_cpy(P.blk[b].b1,   b1_s   + b * 128, 128, tid);
        stage_cpy(P.blk[b].b2,   b2_s   + b * 32,  32,  tid);
        stage_cpy(P.blk[b].lnag, lnag_s + b * 32,  32,  tid);
        stage_cpy(P.blk[b].lnab, lnab_s + b * 32,  32,  tid);
        stage_cpy(P.blk[b].lnbg, lnbg_s + b * 32,  32,  tid);
        stage_cpy(P.blk[b].lnbb, lnbb_s + b * 32,  32,  tid);
    }
    stage_cpy(P.wf,   wf_s,   320, tid);
    stage_cpy(P.bf,   bf_s,   320, tid);
    stage_cpy(P.cls,  cls_s,  32,  tid);
    stage_cpy(P.clfw, clfw_s, 32,  tid);
    stage_cpy(P.clfb, clfb_s, 1,   tid);
    __syncthreads();

    WP W[2];
#pragma unroll
    for (int b = 0; b < 2; ++b) {
        W[b].qkvw = qkvw_s + b * 3072; W[b].outw = outw_s + b * 1024;
        W[b].w1 = w1_s + b * 4096;     W[b].w2 = w2_s + b * 4096;
        W[b].qkvb = qkvb_s + b * 96;   W[b].outb = outb_s + b * 32;
        W[b].b1 = b1_s + b * 128;      W[b].b2 = b2_s + b * 32;
        W[b].lnag = lnag_s + b * 32;   W[b].lnab = lnab_s + b * 32;
        W[b].lnbg = lnbg_s + b * 32;   W[b].lnbb = lnbb_s + b * 32;
    }

    const int wid = tid >> 5, lane = tid & 31;
    const int pairid = wid >> 1;        // group slot 0..3
    const int pw     = wid & 1;         // wave index within pair
    const int id     = pw * 32 + lane;  // 0..63 within pair
    char* gp = sm + OFF_GRP + pairid * GSTRIDE;
    float* ACT = (float*)(gp + GOFF_ACT);
    h16*   QKV = (h16*)(gp + GOFF_QKV);
    h16*   CTX = (h16*)(gp + GOFF_CTX);

    // block-uniform outer loop: each block handles NPAIR groups per iteration
    for (int gb = blockIdx.x; gb < P.ngblocks; gb += gridDim.x) {
        const int g = gb * NPAIR + pairid;
        const bool gvalid = (g < P.ngroups);
        const long base = (long)g * NB;
        int nvalid = gvalid ? (P.Btot - (int)base) : 0;
        if (nvalid > NB) nvalid = NB;

        // ---- build tokens: row t = bl*11 + s ; f32 into ACT ----
        if (gvalid) {
            for (int t = id; t < TOK; t += 64) {
                int bl = t / SEQ, s = t - bl * SEQ;
                long b = base + bl;
                if (b >= (long)P.Btot) continue;
                float* row = ACT + t * DMODEL;
                if (s == 0) {
#pragma unroll
                    for (int d = 0; d < DMODEL; ++d) row[d] = cls_s[d];
                } else {
                    float xv = P.x[b * 10 + (s - 1)];
                    const float* wfr = wf_s + (s - 1) * DMODEL;
                    const float* bfr = bf_s + (s - 1) * DMODEL;
#pragma unroll
                    for (int d = 0; d < DMODEL; ++d) row[d] = xv * wfr[d] + bfr[d];
                }
            }
        }
        __syncthreads();

        run_block(ACT, QKV, CTX, W[0], lane, pw, gvalid, nvalid);
        run_block(ACT, QKV, CTX, W[1], lane, pw, gvalid, nvalid);

        // ---- classifier on CLS token ----
        if (gvalid && id < NB) {
            long b = base + id;
            if (b < (long)P.Btot) {
                const float* row = ACT + id * SEQ * DMODEL;
                float acc = clfb_s[0];
#pragma unroll
                for (int d = 0; d < DMODEL; ++d) acc += row[d] * clfw_s[d];
                P.out[b] = acc;
            }
        }
        __syncthreads();   // protect ACT reuse across outer iterations
    }
}

extern "C" void kernel_launch(void* const* d_in, const int* in_sizes, int n_in,
                              void* d_out, int out_size, void* d_ws, size_t ws_size,
                              hipStream_t stream) {
    (void)n_in; (void)out_size; (void)d_ws; (void)ws_size;
    TParams P;
    P.x   = (const float*)d_in[0];
    P.wf  = (const float*)d_in[1];
    P.bf  = (const float*)d_in[2];
    P.cls = (const float*)d_in[3];
    int o = 4;
    for (int b = 0; b < 2; ++b) {
        P.blk[b].qkvw = (const float*)d_in[o + 0];
        P.blk[b].qkvb = (const float*)d_in[o + 1];
        P.blk[b].outw = (const float*)d_in[o + 2];
        P.blk[b].outb = (const float*)d_in[o + 3];
        P.blk[b].lnag = (const float*)d_in[o + 4];
        P.blk[b].lnab = (const float*)d_in[o + 5];
        P.blk[b].w1   = (const float*)d_in[o + 6];
        P.blk[b].b1   = (const float*)d_in[o + 7];
        P.blk[b].w2   = (const float*)d_in[o + 8];
        P.blk[b].b2   = (const float*)d_in[o + 9];
        P.blk[b].lnbg = (const float*)d_in[o + 10];
        P.blk[b].lnbb = (const float*)d_in[o + 11];
        o += 12;
    }
    P.clfw = (const float*)d_in[28];
    P.clfb = (const float*)d_in[29];
    P.out  = (float*)d_out;
    P.Btot = in_sizes[0] / 10;
    P.ngroups  = (P.Btot + NB - 1) / NB;
    P.ngblocks = (P.ngroups + NPAIR - 1) / NPAIR;

    (void)hipFuncSetAttribute((const void*)fttrans_fused_kernel,
                              hipFuncAttributeMaxDynamicSharedMemorySize, SMEM_BYTES);
    int grid = P.ngblocks;
    if (grid > 1024) grid = 1024;
    fttrans_fused_kernel<<<grid, 256, SMEM_BYTES, stream>>>(P);
}
